// FlashMergeAttentionRope_69346541961839
// MI455X (gfx1250) — compile-verified, matrix operand order checked by reading, not run on hardware
//
#include <hip/hip_runtime.h>

// ---------------------------------------------------------------------------
// Problem constants (from the reference)
// ---------------------------------------------------------------------------
#define BB    4
#define NN    4096
#define CC    1024          // DIM
#define HH    16            // heads
#define DD    64            // head dim
#define C3    3072          // 3*DIM
#define ND    1024          // num_dst  (32*32)
#define NA    3072          // num a tokens

typedef unsigned short bf16_t;                      // raw bf16 bits
typedef __attribute__((ext_vector_type(16))) __bf16 v16bf;
typedef __attribute__((ext_vector_type(8)))  float  v8f;

struct alignas(16) U128 { unsigned int x, y, z, w; };
union Frag { v16bf v; U128 q[2]; };

__device__ __forceinline__ bf16_t f2bf(float f) {
    unsigned int u = __float_as_uint(f);
    u += 0x7FFFu + ((u >> 16) & 1u);                // round-to-nearest-even
    return (bf16_t)(u >> 16);
}
__device__ __forceinline__ float bf2f(bf16_t h) {
    return __uint_as_float(((unsigned int)h) << 16);
}

// Load a 16-bit WMMA A/B fragment (16x32 row-major source, contiguous rows).
// lane L owns row L%16; K elements at (L/16)*8 and 16+(L/16)*8.
__device__ __forceinline__ void load_frag(Frag& f, const bf16_t* rowptr, int k0, int lane) {
    int kb = k0 + ((lane >> 4) << 3);
    f.q[0] = *(const U128*)(rowptr + kb);
    f.q[1] = *(const U128*)(rowptr + kb + 16);
}

__device__ __forceinline__ v8f wmma_bf16(const Frag& a, const Frag& b, v8f c) {
    return __builtin_amdgcn_wmma_f32_16x16x32_bf16(false, a.v, false, b.v,
                                                   (short)0, c, false, false);
}

// a-rank i -> token position (even rows keep odd cols, odd rows keep all cols)
__device__ __forceinline__ int a_pos(int i) {
    int p = i / 96, rem = i % 96;
    return (rem < 32) ? ((2 * p) * 64 + 2 * rem + 1)
                      : ((2 * p + 1) * 64 + (rem - 32));
}
// dst-rank d -> token position (top-left of each 2x2 block)
__device__ __forceinline__ int b_pos(int d) {
    return (d >> 5) * 128 + (d & 31) * 2;
}

// ---------------------------------------------------------------------------
// 0) elementwise fp32 -> bf16 cast (weights)
// ---------------------------------------------------------------------------
__global__ void cast_f32_bf16(const float* __restrict__ src, bf16_t* __restrict__ dst, int n) {
    for (int i = blockIdx.x * blockDim.x + threadIdx.x; i < n; i += gridDim.x * blockDim.x)
        dst[i] = f2bf(src[i]);
}

// ---------------------------------------------------------------------------
// 1) per-token L2 norm; emit xh (bf16) and mh (normalized bf16). One block/token.
// ---------------------------------------------------------------------------
__global__ void norm_cast_kernel(const float* __restrict__ x,
                                 bf16_t* __restrict__ xh, bf16_t* __restrict__ mh) {
    __shared__ float sred[8];
    const int t = blockIdx.x;
    const float* row = x + (size_t)t * CC;
    int lane = threadIdx.x & 31, wid = threadIdx.x >> 5;
    float ss = 0.f;
    for (int c = threadIdx.x; c < CC; c += 256) { float v = row[c]; ss += v * v; }
    for (int off = 1; off < 32; off <<= 1) ss += __shfl_xor(ss, off, 32);
    if (lane == 0) sred[wid] = ss;
    __syncthreads();
    float tot = 0.f;
    for (int w = 0; w < 8; ++w) tot += sred[w];
    float inv = rsqrtf(tot);
    for (int c = threadIdx.x; c < CC; c += 256) {
        float v = row[c];
        xh[(size_t)t * CC + c] = f2bf(v);
        mh[(size_t)t * CC + c] = f2bf(v * inv);
    }
}

// ---------------------------------------------------------------------------
// 2) generic bf16 GEMM:  C[m][n] = sum_k A[m][k] * Bm[n][k]  (+bias)
//    one wave per 16x64 output strip; k-loop software-pipelined (double
//    buffered fragments) so global_load_b128 clauses overlap the WMMAs.
// ---------------------------------------------------------------------------
__global__ void gemm_bf16_nt(const bf16_t* __restrict__ A, const bf16_t* __restrict__ Bm,
                             const float* __restrict__ bias,
                             float* __restrict__ Cf, bf16_t* __restrict__ Cbf,
                             int M, int N, int K) {
    int wave = (blockIdx.x * blockDim.x + threadIdx.x) >> 5;
    int lane = threadIdx.x & 31;
    int tiles_n = N >> 6;
    int total = (M >> 4) * tiles_n;
    if (wave >= total) return;
    int tm = wave / tiles_n, tn = wave % tiles_n;
    const bf16_t* arow = A + (size_t)(tm * 16 + (lane & 15)) * K;
    const bf16_t* brow[4];
    for (int j = 0; j < 4; ++j)
        brow[j] = Bm + (size_t)(tn * 64 + j * 16 + (lane & 15)) * K;

    v8f acc[4] = {};
    Frag a_cur, b_cur[4];
    load_frag(a_cur, arow, 0, lane);
    for (int j = 0; j < 4; ++j) load_frag(b_cur[j], brow[j], 0, lane);

    for (int k0 = 0; k0 + 32 < K; k0 += 32) {
        Frag a_nxt, b_nxt[4];
        // issue next tile's loads first -> one clause, partial loadcnt waits
        load_frag(a_nxt, arow, k0 + 32, lane);
        for (int j = 0; j < 4; ++j) load_frag(b_nxt[j], brow[j], k0 + 32, lane);
        __builtin_prefetch(arow + k0 + 256, 0, 1);     // global_prefetch_b8
        for (int j = 0; j < 4; ++j) acc[j] = wmma_bf16(a_cur, b_cur[j], acc[j]);
        a_cur = a_nxt;
        for (int j = 0; j < 4; ++j) b_cur[j] = b_nxt[j];
    }
    for (int j = 0; j < 4; ++j) acc[j] = wmma_bf16(a_cur, b_cur[j], acc[j]);

    int half = lane >> 4;
    for (int j = 0; j < 4; ++j) {
        int col = tn * 64 + j * 16 + (lane & 15);
        float bi = bias ? bias[col] : 0.f;
        for (int v = 0; v < 8; ++v) {
            int r = tm * 16 + half * 8 + v;
            float val = acc[j][v] + bi;
            if (Cf)  Cf[(size_t)r * N + col] = val;
            if (Cbf) Cbf[(size_t)r * N + col] = f2bf(val);
        }
    }
}

// ---------------------------------------------------------------------------
// 3) scores = m_a . m_b^T with streaming row-argmax -> node_idx  (B x NA)
//    one wave per 16 a-rows; 64 dst tiles processed 4 at a time, k-loop
//    double buffered like the GEMM.
// ---------------------------------------------------------------------------
__global__ void score_argmax_kernel(const bf16_t* __restrict__ Mh, int* __restrict__ node_idx) {
    int wave = (blockIdx.x * blockDim.x + threadIdx.x) >> 5;   // B * 192 waves
    int lane = threadIdx.x & 31;
    int b = wave / (NA / 16), at = wave % (NA / 16);
    int i = at * 16 + (lane & 15);
    const bf16_t* arow = Mh + ((size_t)b * NN + a_pos(i)) * CC;
    float bestv[8]; int besti[8];
    for (int v = 0; v < 8; ++v) { bestv[v] = -3.4e38f; besti[v] = 0; }
    for (int nt4 = 0; nt4 < 64; nt4 += 4) {
        const bf16_t* brow[4];
        for (int j = 0; j < 4; ++j) {
            int d = (nt4 + j) * 16 + (lane & 15);
            brow[j] = Mh + ((size_t)b * NN + b_pos(d)) * CC;
        }
        v8f acc[4] = {};
        Frag a_cur, b_cur[4];
        load_frag(a_cur, arow, 0, lane);
        for (int j = 0; j < 4; ++j) load_frag(b_cur[j], brow[j], 0, lane);
        for (int k0 = 0; k0 + 32 < CC; k0 += 32) {
            Frag a_nxt, b_nxt[4];
            load_frag(a_nxt, arow, k0 + 32, lane);
            for (int j = 0; j < 4; ++j) load_frag(b_nxt[j], brow[j], k0 + 32, lane);
            for (int j = 0; j < 4; ++j) acc[j] = wmma_bf16(a_cur, b_cur[j], acc[j]);
            a_cur = a_nxt;
            for (int j = 0; j < 4; ++j) b_cur[j] = b_nxt[j];
        }
        for (int j = 0; j < 4; ++j) acc[j] = wmma_bf16(a_cur, b_cur[j], acc[j]);

        for (int j = 0; j < 4; ++j) {
            int col = (nt4 + j) * 16 + (lane & 15);
            for (int v = 0; v < 8; ++v)
                if (acc[j][v] > bestv[v]) { bestv[v] = acc[j][v]; besti[v] = col; }
        }
    }
    // argmax reduce across the 16 lanes holding each row (prefer lower index on tie)
    for (int off = 1; off < 16; off <<= 1) {
        for (int v = 0; v < 8; ++v) {
            float ov = __shfl_xor(bestv[v], off, 32);
            int   oi = __shfl_xor(besti[v], off, 32);
            if (ov > bestv[v] || (ov == bestv[v] && oi < besti[v])) { bestv[v] = ov; besti[v] = oi; }
        }
    }
    if ((lane & 15) == 0) {
        int half = lane >> 4;
        for (int v = 0; v < 8; ++v)
            node_idx[b * NA + at * 16 + half * 8 + v] = besti[v];
    }
}

// ---------------------------------------------------------------------------
// 4) merge: counts + fp32 segment-sum + mean; emit Q,K (b,h,n,d), V^T (b,h,d,n)
// ---------------------------------------------------------------------------
__global__ void zero_i32(int* p, int n) {
    for (int i = blockIdx.x * blockDim.x + threadIdx.x; i < n; i += gridDim.x * blockDim.x) p[i] = 0;
}
__global__ void count_kernel(const int* __restrict__ node_idx, int* __restrict__ cnt) {
    int i = blockIdx.x * blockDim.x + threadIdx.x;
    if (i >= BB * NA) return;
    int b = i / NA;
    atomicAdd(&cnt[b * ND + node_idx[i]], 1);
}
__global__ void merge_init_kernel(const bf16_t* __restrict__ qkvh, float* __restrict__ accm) {
    size_t i = (size_t)blockIdx.x * blockDim.x + threadIdx.x;   // B*ND*C3
    if (i >= (size_t)BB * ND * C3) return;
    int c = i % C3; int d = (i / C3) % ND; int b = i / ((size_t)ND * C3);
    accm[i] = bf2f(qkvh[((size_t)b * NN + b_pos(d)) * C3 + c]);
}
__global__ void merge_accum_kernel(const bf16_t* __restrict__ qkvh,
                                   const int* __restrict__ node_idx,
                                   float* __restrict__ accm) {
    size_t t = (size_t)blockIdx.x * blockDim.x + threadIdx.x;   // B*NA*C3
    if (t >= (size_t)BB * NA * C3) return;
    int c = t % C3; int i = (t / C3) % NA; int b = t / ((size_t)NA * C3);
    int d = node_idx[b * NA + i];
    float v = bf2f(qkvh[((size_t)b * NN + a_pos(i)) * C3 + c]);
    atomicAdd(&accm[((size_t)b * ND + d) * C3 + c], v);
}
__global__ void merge_final_kernel(const float* __restrict__ accm, const int* __restrict__ cnt,
                                   bf16_t* __restrict__ qh, bf16_t* __restrict__ kh,
                                   bf16_t* __restrict__ vTh) {
    size_t i = (size_t)blockIdx.x * blockDim.x + threadIdx.x;   // B*ND*C3
    if (i >= (size_t)BB * ND * C3) return;
    int c = i % C3; int n = (i / C3) % ND; int b = i / ((size_t)ND * C3);
    float val = accm[i] / (1.0f + (float)cnt[b * ND + n]);
    int three = c / CC, h = (c % CC) / DD, dd = c % DD;
    bf16_t bv = f2bf(val);
    size_t bh = (size_t)(b * HH + h);
    if (three == 0)      qh [(bh * ND + n) * DD + dd] = bv;
    else if (three == 1) kh [(bh * ND + n) * DD + dd] = bv;
    else                 vTh[(bh * DD + dd) * ND + n] = bv;
}

// ---------------------------------------------------------------------------
// 5) flash attention: wave owns 16 q rows of one (b,h); online softmax,
//    QK^T and P*V via WMMA; P re-laid-out through LDS into an A fragment.
//    K fragments issue as one clause; V fragments are hoisted before the
//    softmax so their latency hides behind QK^T WMMAs + exp/shuffle work.
// ---------------------------------------------------------------------------
__global__ void __launch_bounds__(128)
flash_attn_kernel(const bf16_t* __restrict__ Q, const bf16_t* __restrict__ K,
                  const bf16_t* __restrict__ Vt, bf16_t* __restrict__ O) {
    __shared__ bf16_t pbuf[4][16][32];
    int wid = threadIdx.x >> 5, lane = threadIdx.x & 31;
    int task = blockIdx.x * 4 + wid;          // b*1024 + h*64 + qt
    int qt = task & 63, h = (task >> 6) & 15, b = task >> 10;
    int half = lane >> 4;
    const float scale = 0.125f;               // 1/sqrt(64)

    const bf16_t* qrow = Q + (((size_t)(b * HH + h)) * ND + qt * 16 + (lane & 15)) * DD;
    Frag qa0, qa1;
    load_frag(qa0, qrow, 0, lane);
    load_frag(qa1, qrow, 32, lane);

    const bf16_t* kbh = K  + ((size_t)(b * HH + h)) * ND * DD;
    const bf16_t* vbh = Vt + ((size_t)(b * HH + h)) * DD * ND;

    v8f acc[4] = {};
    float m[8], l[8];
    for (int v = 0; v < 8; ++v) { m[v] = -3.4e38f; l[v] = 0.f; }

    for (int kt = 0; kt < ND; kt += 32) {
        // ---- issue all K-tile fragments as one clause
        const bf16_t* kr0 = kbh + (size_t)(kt + (lane & 15)) * DD;
        const bf16_t* kr1 = kbh + (size_t)(kt + 16 + (lane & 15)) * DD;
        Frag kb00, kb01, kb10, kb11;
        load_frag(kb00, kr0, 0,  lane);
        load_frag(kb01, kr0, 32, lane);
        load_frag(kb10, kr1, 0,  lane);
        load_frag(kb11, kr1, 32, lane);
        // ---- V fragments early: independent of S, overlap with softmax
        Frag vb[4];
        for (int ct = 0; ct < 4; ++ct)
            load_frag(vb[ct], vbh + (size_t)(ct * 16 + (lane & 15)) * ND, kt, lane);

        // ---- S = Q K^T (two 16x16 tiles, K-dim = 64 = 2 wmma each)
        v8f s0 = {}, s1 = {};
        s0 = wmma_bf16(qa0, kb00, s0);
        s0 = wmma_bf16(qa1, kb01, s0);
        s1 = wmma_bf16(qa0, kb10, s1);
        s1 = wmma_bf16(qa1, kb11, s1);

        // ---- online softmax update over these 32 keys
        float rmax[8];
        for (int v = 0; v < 8; ++v)
            rmax[v] = fmaxf(s0[v], s1[v]) * scale;
        for (int off = 1; off < 16; off <<= 1)
            for (int v = 0; v < 8; ++v)
                rmax[v] = fmaxf(rmax[v], __shfl_xor(rmax[v], off, 32));
        float rsum[8];
        for (int v = 0; v < 8; ++v) {
            float nm = fmaxf(m[v], rmax[v]);
            float corr = __expf(m[v] - nm);
            m[v] = nm;
            l[v] *= corr;
            for (int j = 0; j < 4; ++j) acc[j][v] *= corr;
            float p0 = __expf(s0[v] * scale - nm);
            float p1 = __expf(s1[v] * scale - nm);
            rsum[v] = p0 + p1;
            pbuf[wid][half * 8 + v][lane & 15]        = f2bf(p0);
            pbuf[wid][half * 8 + v][16 + (lane & 15)] = f2bf(p1);
        }
        for (int off = 1; off < 16; off <<= 1)
            for (int v = 0; v < 8; ++v)
                rsum[v] += __shfl_xor(rsum[v], off, 32);
        for (int v = 0; v < 8; ++v) l[v] += rsum[v];

        __syncthreads();   // order LDS p-stores before fragment reads (uniform loop)

        // ---- O += P (16x32) * V (32x64)
        Frag pa;
        pa.q[0] = *(const U128*)&pbuf[wid][lane & 15][half * 8];
        pa.q[1] = *(const U128*)&pbuf[wid][lane & 15][16 + half * 8];
        for (int ct = 0; ct < 4; ++ct)
            acc[ct] = wmma_bf16(pa, vb[ct], acc[ct]);
        __syncthreads();
    }

    // ---- epilogue: O[b][n][h*64+d] = acc / l
    for (int v = 0; v < 8; ++v) {
        float inv = 1.0f / l[v];
        int r = qt * 16 + half * 8 + v;
        for (int ct = 0; ct < 4; ++ct) {
            int col = h * DD + ct * 16 + (lane & 15);
            O[((size_t)b * ND + r) * CC + col] = f2bf(acc[ct][v] * inv);
        }
    }
}

// ---------------------------------------------------------------------------
// 6) unmerge scatter: every output token copies its merged row (closed-form map)
// ---------------------------------------------------------------------------
__global__ void unmerge_kernel(const float* __restrict__ y, const int* __restrict__ node_idx,
                               float* __restrict__ out) {
    int bt = blockIdx.x;                 // B*NN blocks
    int b = bt >> 12, t = bt & 4095;
    int r = t >> 6, c = t & 63;
    int g;
    if (((r & 1) == 0) && ((c & 1) == 0)) {
        g = (r >> 1) * 32 + (c >> 1);                       // dst token
    } else {
        int nb = ((r + 1) >> 1) * 32 + (((r & 1) == 0) ? ((c + 1) >> 1) : 0);
        g = node_idx[b * NA + (t - nb)];                    // merged-into dst
    }
    const float4* src = (const float4*)(y  + ((size_t)b * ND + g) * CC);
    float4*       dst = (float4*)      (out + ((size_t)b * NN + t) * CC);
    int i = threadIdx.x;                 // 256 threads * float4 = 1024 floats
    dst[i] = src[i];
}

// ---------------------------------------------------------------------------
// host launcher
// ---------------------------------------------------------------------------
extern "C" void kernel_launch(void* const* d_in, const int* in_sizes, int n_in,
                              void* d_out, int out_size, void* d_ws, size_t ws_size,
                              hipStream_t stream) {
    const float* x      = (const float*)d_in[0];
    const float* W_qkv  = (const float*)d_in[1];
    const float* W_proj = (const float*)d_in[2];
    const float* b_proj = (const float*)d_in[3];
    float* out = (float*)d_out;

    // carve workspace
    char* p = (char*)d_ws;
    auto alloc = [&](size_t bytes) -> void* {
        void* r = (void*)p; p += (bytes + 255) & ~(size_t)255; return r;
    };
    bf16_t* xh     = (bf16_t*)alloc((size_t)BB * NN * CC * 2);
    bf16_t* mh     = (bf16_t*)alloc((size_t)BB * NN * CC * 2);
    bf16_t* wqkvh  = (bf16_t*)alloc((size_t)C3 * CC * 2);
    bf16_t* wprojh = (bf16_t*)alloc((size_t)CC * CC * 2);
    bf16_t* qkvh   = (bf16_t*)alloc((size_t)BB * NN * C3 * 2);
    int*    nodei  = (int*)   alloc((size_t)BB * NA * 4);
    int*    cnt    = (int*)   alloc((size_t)BB * ND * 4);
    float*  accm   = (float*) alloc((size_t)BB * ND * C3 * 4);
    bf16_t* qh     = (bf16_t*)alloc((size_t)BB * HH * ND * DD * 2);
    bf16_t* kh     = (bf16_t*)alloc((size_t)BB * HH * ND * DD * 2);
    bf16_t* vTh    = (bf16_t*)alloc((size_t)BB * HH * DD * ND * 2);
    bf16_t* oh     = (bf16_t*)alloc((size_t)BB * ND * CC * 2);
    float*  y      = (float*) alloc((size_t)BB * ND * CC * 4);

    // 0) weight casts
    cast_f32_bf16<<<4096, 256, 0, stream>>>(W_qkv,  wqkvh,  C3 * CC);
    cast_f32_bf16<<<2048, 256, 0, stream>>>(W_proj, wprojh, CC * CC);

    // 1) token norms + casts
    norm_cast_kernel<<<BB * NN, 256, 0, stream>>>(x, xh, mh);

    // 2) QKV GEMM: (16384 x 1024) x (1024 x 3072)  -> bf16
    {
        int waves = (BB * NN / 16) * (C3 / 64);      // 49152
        gemm_bf16_nt<<<waves / 8, 256, 0, stream>>>(xh, wqkvh, nullptr,
                                                    nullptr, qkvh, BB * NN, C3, CC);
    }

    // 3) scores + argmax -> node_idx
    score_argmax_kernel<<<(BB * (NA / 16)) / 8, 256, 0, stream>>>(mh, nodei);

    // 4) merge
    zero_i32<<<16, 256, 0, stream>>>(cnt, BB * ND);
    count_kernel<<<(BB * NA + 255) / 256, 256, 0, stream>>>(nodei, cnt);
    {
        size_t n_init = (size_t)BB * ND * C3;        // 12.58M
        merge_init_kernel<<<(unsigned)((n_init + 255) / 256), 256, 0, stream>>>(qkvh, accm);
        size_t n_acc = (size_t)BB * NA * C3;         // 37.75M
        merge_accum_kernel<<<(unsigned)((n_acc + 255) / 256), 256, 0, stream>>>(qkvh, nodei, accm);
        merge_final_kernel<<<(unsigned)((n_init + 255) / 256), 256, 0, stream>>>(accm, cnt, qh, kh, vTh);
    }

    // 5) attention: 4096 wave-tasks, 4 waves/block
    flash_attn_kernel<<<(BB * HH * (ND / 16)) / 4, 128, 0, stream>>>(qh, kh, vTh, oh);

    // 6) projection GEMM + bias: (4096 x 1024) x (1024 x 1024) -> f32
    {
        int waves = (BB * ND / 16) * (CC / 64);      // 4096
        gemm_bf16_nt<<<waves / 8, 256, 0, stream>>>(oh, wprojh, b_proj,
                                                    y, nullptr, BB * ND, CC, CC);
    }

    // 7) unmerge scatter
    unmerge_kernel<<<BB * NN, 256, 0, stream>>>(y, nodei, out);
}
